// GQattention_56564719288388
// MI455X (gfx1250) — matrix operand order repaired and checked by reading, not compiled, not run on hardware
//
#include <hip/hip_runtime.h>

// ---------------------------------------------------------------------------
// GQA-without-softmax attention block for MI455X (gfx1250, wave32, WMMA).
// Pipeline:
//   1) proj<ROPE>(x,Wq)->q_bf16   2) proj<PLAIN>(x,Wk)->k_bf16
//   3) proj<ROPE>(x,Wv)->v_bf16   4) fused attn (QK^T/sqrt(d)) V -> attn_bf16
//   5) proj<F32OUT>(attn,Wo)+b -> d_out (fp32)
// All matmuls: V_WMMA_F32_16X16X32_BF16, f32 accumulate.
// K tiles: double-buffered TDM (tensor_load_to_lds) prefetch, TENSORcnt.
// Q tile:  async global->LDS b128, ASYNCcnt.
// Double buffers selected by offset arithmetic (not pointer arrays) so
// address-space inference keeps all fragment reads on the DS path.
// ---------------------------------------------------------------------------

typedef __attribute__((ext_vector_type(16))) __bf16 v16bf;
typedef __attribute__((ext_vector_type(8)))  __bf16 v8bf;
typedef __attribute__((ext_vector_type(8)))  float  v8f;
typedef __attribute__((ext_vector_type(4)))  int    v4i_;
typedef __attribute__((ext_vector_type(4)))  unsigned int v4u_;
typedef __attribute__((ext_vector_type(8)))  int    v8i_;

#define EMB_   2048
#define SEQ_   2048
#define BATCH_ 4
#define NH_    16
#define HD_    128     // head dim

#if defined(__AMDGCN__) && __has_builtin(__builtin_amdgcn_tensor_load_to_lds) && \
    __has_builtin(__builtin_amdgcn_s_wait_tensorcnt)
#define USE_TDM 1
#warning "CDNA5 probe: TDM tensor_load_to_lds path ENABLED"
#else
#define USE_TDM 0
#endif

#if defined(__AMDGCN__) && __has_builtin(__builtin_amdgcn_global_load_async_to_lds_b128) && \
    __has_builtin(__builtin_amdgcn_s_wait_asynccnt)
#define USE_ASYNC 1
#warning "CDNA5 probe: async global->LDS path ENABLED"
#else
#define USE_ASYNC 0
#endif

__device__ __forceinline__ v8f zero_v8f() {
    v8f z = {0.f, 0.f, 0.f, 0.f, 0.f, 0.f, 0.f, 0.f};
    return z;
}

__device__ __forceinline__ v8f wmma_bf16(v16bf a, v16bf b, v8f c) {
    // (neg_a, A, neg_b, B, c_mod, C, reuse_a, reuse_b)
    return __builtin_amdgcn_wmma_f32_16x16x32_bf16(false, a, false, b,
                                                   (short)0, c, false, false);
}

// A fragment (16x32, bf16): lane l -> row = l%16, K = 8*(l/16)+0..7 and +16..23.
__device__ __forceinline__ v16bf lds_a_frag(const __bf16* base, int row0,
                                            int ld, int k0, int lane) {
    const int row = row0 + (lane & 15);
    const int kk  = k0 + ((lane >> 4) << 3);
    const __bf16* p = base + row * ld + kk;
    v8bf lo = *(const v8bf*)(p);
    v8bf hi = *(const v8bf*)(p + 16);
    return __builtin_shufflevector(lo, hi, 0,1,2,3,4,5,6,7,8,9,10,11,12,13,14,15);
}

// B fragment (32x16, bf16) from an LDS tile stored [N][K]:
// lane l -> col = l%16, K = 8*(l/16)+0..7 and +16..23.
__device__ __forceinline__ v16bf lds_b_frag(const __bf16* base, int col0,
                                            int ld, int k0, int lane) {
    const int col = col0 + (lane & 15);
    const int kk  = k0 + ((lane >> 4) << 3);
    const __bf16* p = base + col * ld + kk;
    v8bf lo = *(const v8bf*)(p);
    v8bf hi = *(const v8bf*)(p + 16);
    return __builtin_shufflevector(lo, hi, 0,1,2,3,4,5,6,7,8,9,10,11,12,13,14,15);
}

#if USE_TDM
// One TDM descriptor moving a 128x128 bf16 tile (row stride = row_stride elems)
// from global memory into LDS. Issued once per wave that executes it.
__device__ __forceinline__ void tdm_load_tile_128x128_bf16(
        const __bf16* gsrc, const __bf16* ldst, unsigned row_stride)
{
    const unsigned long long ga =
        (unsigned long long)(uintptr_t)gsrc;
    const unsigned lds_off = (unsigned)(uintptr_t)
        (__attribute__((address_space(3))) const __bf16*)ldst;

    // D# group 0: count=1 | lds_addr | global_addr[31:0] |
    //             global_addr[56:32] + type=2 (bits 127:126)
    v4u_ g0 = { 1u,
                lds_off,
                (unsigned)(ga & 0xffffffffull),
                (unsigned)((ga >> 32) & 0x1ffffffull) | 0x80000000u };
    // D# group 1: wg_mask=0, data_size=2B(code 1); tensor_dim0=128,
    //             tensor_dim1=128, tile_dim0=128, tile_dim1=128,
    //             tensor_dim0_stride=row_stride, dim1_stride unused (2D).
    v8i_ g1 = { (int)0x00010000u,       // [17:16]=1 -> 2-byte elements
                (int)(128u << 16),      // tensor_dim0[15:0] @ bits 63:48
                (int)(128u << 16),      // dim0 hi=0, tensor_dim1[15:0] @ 95:80
                (int)(128u << 16),      // dim1 hi=0, tile_dim0 @ 127:112
                128,                    // tile_dim1 @ 143:128
                (int)row_stride,        // tensor_dim0_stride[31:0] @ 191:160
                0, 0 };
    v4i_ gz = { 0, 0, 0, 0 };
#if __clang_major__ >= 23
    v8i_ gz8 = { 0, 0, 0, 0, 0, 0, 0, 0 };
    __builtin_amdgcn_tensor_load_to_lds(g0, g1, gz, gz, gz8, 0);
#else
    __builtin_amdgcn_tensor_load_to_lds(g0, g1, gz, gz, 0);
#endif
}
#endif // USE_TDM

// ---------------------------------------------------------------------------
// Projection GEMM: C[M=8192, N=2048] = A[M,2048] * W^T (+bias), epilogue modes:
//   MODE 0: bf16 store (K proj)
//   MODE 1: bias + full-dim RoPE, bf16 store (Q,V proj)
//   MODE 2: bias, fp32 store (output proj)
// Block = 256 thr (8 waves), tile 128x128, K-step 32. Wave tile 64x32.
// ---------------------------------------------------------------------------
template <int MODE, typename AT>
__global__ __launch_bounds__(256)
void proj_kernel(const AT* __restrict__ A, const float* __restrict__ W,
                 const float* __restrict__ bias, void* __restrict__ OutP)
{
    __shared__ __bf16 As[128 * 32];   // [M][K]
    __shared__ __bf16 Bs[128 * 32];   // [N][K]  (W rows == B columns)

    const int tid  = threadIdx.x;
    const int lane = tid & 31;
    const int wave = tid >> 5;
    const int wm   = (wave >> 2) * 64;   // wave M offset in tile
    const int wn   = (wave & 3) * 32;    // wave N offset in tile
    const int m0   = blockIdx.y * 128;
    const int n0   = blockIdx.x * 128;

    v8f acc[4][2];
#pragma unroll
    for (int mi = 0; mi < 4; ++mi)
#pragma unroll
        for (int ni = 0; ni < 2; ++ni) acc[mi][ni] = zero_v8f();

    const int lrow = tid >> 1;          // 0..127
    const int lcol = (tid & 1) * 16;    // 0 or 16

    for (int k0 = 0; k0 < EMB_; k0 += 32) {
        const AT*    asrc = A + (size_t)(m0 + lrow) * EMB_ + k0 + lcol;
        const float* bsrc = W + (size_t)(n0 + lrow) * EMB_ + k0 + lcol;
        __bf16* adst = As + lrow * 32 + lcol;
        __bf16* bdst = Bs + lrow * 32 + lcol;
#pragma unroll
        for (int i = 0; i < 16; ++i) adst[i] = (__bf16)(float)asrc[i];
#pragma unroll
        for (int i = 0; i < 16; ++i) bdst[i] = (__bf16)bsrc[i];
        __builtin_prefetch(asrc + 32, 0, 1);   // global_prefetch_b8
        __builtin_prefetch(bsrc + 32, 0, 1);
        __syncthreads();

        v16bf bf0 = lds_b_frag(Bs, wn,      32, 0, lane);
        v16bf bf1 = lds_b_frag(Bs, wn + 16, 32, 0, lane);
#pragma unroll
        for (int mi = 0; mi < 4; ++mi) {
            v16bf af = lds_a_frag(As, wm + mi * 16, 32, 0, lane);
            acc[mi][0] = wmma_bf16(af, bf0, acc[mi][0]);
            acc[mi][1] = wmma_bf16(af, bf1, acc[mi][1]);
        }
        __syncthreads();
    }

    // Epilogue. C layout: VGPR g, lane l -> row = g + 8*(l/16), col = l%16.
#pragma unroll
    for (int mi = 0; mi < 4; ++mi) {
#pragma unroll
        for (int ni = 0; ni < 2; ++ni) {
#pragma unroll
            for (int g = 0; g < 8; ++g) {
                const int row = m0 + wm + mi * 16 + g + ((lane >> 4) << 3);
                const int col = n0 + wn + ni * 16 + (lane & 15);
                float c = acc[mi][ni][g] + bias[col];
                if (MODE == 1) {
                    // RoPE pair (2p,2p+1) = adjacent N cols = adjacent lanes.
                    float other = __shfl_xor(c, 1, 32);
                    const int   p  = col >> 1;                 // 0..1023
                    const float th = (p < (EMB_ / 4)) ? 1.0f : 1e-4f;
                    const float ang = th * (float)((row % SEQ_) + 1);
                    float sn, cs;
                    __sincosf(ang, &sn, &cs);
                    // even col holds x0: r0 = x0*sin - x1*cos
                    // odd  col holds x1: r1 = x0*cos + x1*sin
                    c = ((col & 1) == 0) ? (c * sn - other * cs)
                                         : (other * cs + c * sn);
                }
                if (MODE == 2) {
                    ((float*)OutP)[(size_t)row * EMB_ + col] = c;
                } else {
                    ((__bf16*)OutP)[(size_t)row * EMB_ + col] = (__bf16)c;
                }
            }
        }
    }
}

// ---------------------------------------------------------------------------
// Fused attention (no softmax): per (b,h), per 128-row Q tile:
//   for each 128-row K/V block: S = Q K^T * 1/sqrt(128) -> bf16 in LDS,
//                               O += S V   (accumulated in registers)
// Dynamic LDS: Q + K0 + K1 + V^T + S tiles (bf16, 32 KB each) = 160 KB.
// K tiles are double-buffered TDM prefetches: the DMA for tile i+1 is issued
// right after the barrier frees its buffer, and only s_wait_tensorcnt 1 is
// needed before consuming tile i (per-wave TDM ops complete in order).
// Q tile staged by async global->LDS b128 (ASYNCcnt).
// ---------------------------------------------------------------------------
__global__ __launch_bounds__(256)
void attn_kernel(const __bf16* __restrict__ Qb, const __bf16* __restrict__ Kb,
                 const __bf16* __restrict__ Vb, __bf16* __restrict__ Ob)
{
    extern __shared__ __bf16 smem[];
    __bf16* qs = smem;                  // [q][d]
    // K buffers: smem + (1 + idx)*16384, selected by OFFSET (not ptr select)
    __bf16* vs = smem + 3 * 16384;      // [d][kpos]  (B-tile [N][K] for S*V)
    __bf16* ss = smem + 4 * 16384;      // [q][kpos]  (A-tile for S*V)

    const int tid  = threadIdx.x;
    const int lane = tid & 31;
    const int wave = tid >> 5;
    const int wm   = (wave >> 2) * 64;
    const int wn   = (wave & 3) * 32;
    const int b    = blockIdx.y >> 4;
    const int h    = blockIdx.y & 15;
    const int q0   = blockIdx.x * 128;
    const size_t base = (size_t)b * SEQ_ * EMB_ + (size_t)h * HD_;

    // Load Q tile (resident for the whole block).
#if USE_ASYNC
    for (int c = tid; c < 128 * 16; c += 256) {
        const int row = c >> 4, d = (c & 15) << 3;
        __builtin_amdgcn_global_load_async_to_lds_b128(
            (__attribute__((address_space(1))) v4i_*)
                (Qb + base + (size_t)(q0 + row) * EMB_ + d),
            (__attribute__((address_space(3))) v4i_*)(qs + row * 128 + d),
            0, 0);
    }
    __builtin_amdgcn_s_wait_asynccnt(0);
#else
    for (int c = tid; c < 128 * 16; c += 256) {
        const int row = c >> 4, d = (c & 15) << 3;
        *(v8bf*)(qs + row * 128 + d) =
            *(const v8bf*)(Qb + base + (size_t)(q0 + row) * EMB_ + d);
    }
#endif

#if USE_TDM
    // Prefetch first K tile before entering the loop.
    if (wave == 0)
        tdm_load_tile_128x128_bf16(Kb + base, smem + 16384, EMB_);
#endif

    v8f oacc[4][2];
#pragma unroll
    for (int mi = 0; mi < 4; ++mi)
#pragma unroll
        for (int ni = 0; ni < 2; ++ni) oacc[mi][ni] = zero_v8f();

    const float scale = 0.08838834764831845f;   // 1/sqrt(128)

    for (int kb = 0; kb < SEQ_; kb += 128) {
        const int idx = (kb >> 7) & 1;
        const __bf16* ks  = smem + (1 + idx) * 16384;        // current K tile
        __bf16*       ksn = smem + (1 + (idx ^ 1)) * 16384;  // next K tile
        __syncthreads();   // previous GEMMs done: ksn and vs reusable

        // --- K tile for NEXT iteration: TDM prefetch into the other buffer ---
#if USE_TDM
        if (wave == 0 && (kb + 128) < SEQ_)
            tdm_load_tile_128x128_bf16(Kb + base + (size_t)(kb + 128) * EMB_,
                                       ksn, EMB_);
#else
        for (int c = tid; c < 128 * 16; c += 256) {
            const int row = c >> 4, d = (c & 15) << 3;
            *(v8bf*)((__bf16*)ks + row * 128 + d) =
                *(const v8bf*)(Kb + base + (size_t)(kb + row) * EMB_ + d);
        }
#endif
        // --- V tile transposed into [d][kpos] (register transpose) ---
        for (int c = tid; c < 128 * 16; c += 256) {
            const int row = c >> 4, d0 = (c & 15) << 3;
            v8bf t = *(const v8bf*)(Vb + base + (size_t)(kb + row) * EMB_ + d0);
#pragma unroll
            for (int i = 0; i < 8; ++i) vs[(d0 + i) * 128 + row] = t[i];
        }
#if USE_TDM
        // Tile kb must have landed; the kb+128 prefetch may stay in flight.
        if (wave == 0) {
            if ((kb + 128) < SEQ_) __builtin_amdgcn_s_wait_tensorcnt(1);
            else                   __builtin_amdgcn_s_wait_tensorcnt(0);
        }
#endif
        __syncthreads();

        // GEMM1: S[128q][128k] = Q * K^T  (K-dim = head dim, 4 steps of 32)
        v8f sacc[4][2];
#pragma unroll
        for (int mi = 0; mi < 4; ++mi)
#pragma unroll
            for (int ni = 0; ni < 2; ++ni) sacc[mi][ni] = zero_v8f();

#pragma unroll
        for (int dk = 0; dk < 128; dk += 32) {
            v16bf bf0 = lds_b_frag(ks, wn,      128, dk, lane);
            v16bf bf1 = lds_b_frag(ks, wn + 16, 128, dk, lane);
#pragma unroll
            for (int mi = 0; mi < 4; ++mi) {
                v16bf af = lds_a_frag(qs, wm + mi * 16, 128, dk, lane);
                sacc[mi][0] = wmma_bf16(af, bf0, sacc[mi][0]);
                sacc[mi][1] = wmma_bf16(af, bf1, sacc[mi][1]);
            }
        }
        // Scale + round scores to bf16 into LDS A-tile.
#pragma unroll
        for (int mi = 0; mi < 4; ++mi)
#pragma unroll
            for (int ni = 0; ni < 2; ++ni)
#pragma unroll
                for (int g = 0; g < 8; ++g) {
                    const int r  = wm + mi * 16 + g + ((lane >> 4) << 3);
                    const int cc = wn + ni * 16 + (lane & 15);
                    ss[r * 128 + cc] = (__bf16)(sacc[mi][ni][g] * scale);
                }
        __syncthreads();

        // GEMM2: O += S * V  (K-dim = key positions, 4 steps of 32)
#pragma unroll
        for (int kk = 0; kk < 128; kk += 32) {
            v16bf bf0 = lds_b_frag(vs, wn,      128, kk, lane);
            v16bf bf1 = lds_b_frag(vs, wn + 16, 128, kk, lane);
#pragma unroll
            for (int mi = 0; mi < 4; ++mi) {
                v16bf af = lds_a_frag(ss, wm + mi * 16, 128, kk, lane);
                oacc[mi][0] = wmma_bf16(af, bf0, oacc[mi][0]);
                oacc[mi][1] = wmma_bf16(af, bf1, oacc[mi][1]);
            }
        }
    }

    // Write O tile (bf16) back to [B,S,H*D] layout.
#pragma unroll
    for (int mi = 0; mi < 4; ++mi)
#pragma unroll
        for (int ni = 0; ni < 2; ++ni)
#pragma unroll
            for (int g = 0; g < 8; ++g) {
                const int r = q0 + wm + mi * 16 + g + ((lane >> 4) << 3);
                const int d = wn + ni * 16 + (lane & 15);
                Ob[base + (size_t)r * EMB_ + d] = (__bf16)oacc[mi][ni][g];
            }
}

// ---------------------------------------------------------------------------
extern "C" void kernel_launch(void* const* d_in, const int* in_sizes, int n_in,
                              void* d_out, int out_size, void* d_ws, size_t ws_size,
                              hipStream_t stream)
{
    const float* x    = (const float*)d_in[0];
    const float* wq_w = (const float*)d_in[1];
    const float* wq_b = (const float*)d_in[2];
    const float* wk_w = (const float*)d_in[3];
    const float* wk_b = (const float*)d_in[4];
    const float* wv_w = (const float*)d_in[5];
    const float* wv_b = (const float*)d_in[6];
    const float* wo_w = (const float*)d_in[7];
    const float* wo_b = (const float*)d_in[8];
    // d_in[9] = cur_pos (0, prefill) -- unused.

    const size_t NBSE = (size_t)BATCH_ * SEQ_ * EMB_;   // 16.78M elems
    __bf16* qb = (__bf16*)d_ws;
    __bf16* kb = qb + NBSE;
    __bf16* vb = kb + NBSE;
    __bf16* ob = vb + NBSE;          // 4 * 33.5 MB = 134 MB of workspace

    (void)hipFuncSetAttribute((const void*)attn_kernel,
                              hipFuncAttributeMaxDynamicSharedMemorySize, 163840);

    dim3 gridP(EMB_ / 128, (BATCH_ * SEQ_) / 128);      // (16, 64)
    dim3 gridA(SEQ_ / 128, BATCH_ * NH_);               // (16, 64)

    proj_kernel<1, float ><<<gridP, 256, 0, stream>>>(x,  wq_w, wq_b, (void*)qb);
    proj_kernel<0, float ><<<gridP, 256, 0, stream>>>(x,  wk_w, wk_b, (void*)kb);
    proj_kernel<1, float ><<<gridP, 256, 0, stream>>>(x,  wv_w, wv_b, (void*)vb);
    attn_kernel<<<gridA, 256, 163840, stream>>>(qb, kb, vb, ob);
    proj_kernel<2, __bf16><<<gridP, 256, 0, stream>>>(ob, wo_w, wo_b, d_out);
}
